// CausalAttention_69784628625809
// MI455X (gfx1250) — matrix operand order repaired and checked by reading, not compiled, not run on hardware
//
#include <hip/hip_runtime.h>
#include <math.h>
#include <stdint.h>

// ---------------------------------------------------------------------------
// Causal attention, MI455X (gfx1250), wave32 WMMA flash-attention + TDM.
//   x:[4,4096,512] f32, Wq/Wk/Wv:[512,128] f32 -> out:[4,4096,128] f32
// Pass 1: Q/K (row-major f16) and V (transposed f16, Vt[d][kv]) = x @ W
// Pass 2: fused causal flash attention; K/Vt tiles staged to LDS by the
//         Tensor Data Mover (tensor_load_to_lds, TENSORcnt double-buffered).
// ---------------------------------------------------------------------------

typedef _Float16     v8h   __attribute__((ext_vector_type(8)));
typedef _Float16     v16h  __attribute__((ext_vector_type(16)));
typedef float        v8f   __attribute__((ext_vector_type(8)));
typedef float        v4f   __attribute__((ext_vector_type(4)));
typedef unsigned int u32x4 __attribute__((ext_vector_type(4)));
typedef int          i32x8 __attribute__((ext_vector_type(8)));
typedef int          i32x4 __attribute__((ext_vector_type(4)));

#define BATCH 4
#define SEQ   4096
#define DIN   512
#define DOUT  128
#define BS    (BATCH * SEQ)

static __device__ __forceinline__ v16h cat8(v8h lo, v8h hi) {
  return __builtin_shufflevector(lo, hi, 0,1,2,3,4,5,6,7,8,9,10,11,12,13,14,15);
}

// LDS byte offset of a __shared__ object (AS3 pointers are 32-bit LDS offsets)
typedef __attribute__((address_space(3))) void lds_void;
static __device__ __forceinline__ uint32_t lds_off(void* p) {
  return (uint32_t)(uintptr_t)(lds_void*)p;
}

// Issue one 2D TDM tile load: tile_d0 x tile_d1 elements of 2 bytes,
// row stride stride0 (elements), into contiguous LDS at lds_byte_addr.
// D# bitfields per CDNA5 ISA §8.3 (Group0) / §8.4 (Group1).
static __device__ __forceinline__ void tdm_load_2d(uint32_t lds_byte_addr,
                                                   const void* gtile,
                                                   uint32_t tensor_d0,
                                                   uint32_t tensor_d1,
                                                   uint32_t tile_d0,
                                                   uint32_t tile_d1,
                                                   uint64_t stride0) {
  const uint64_t ga = (uint64_t)(uintptr_t)gtile;
  u32x4 g0;
  g0[0] = 1u;                                           // count=1, user mode
  g0[1] = lds_byte_addr;                                // lds_addr
  g0[2] = (uint32_t)(ga & 0xffffffffu);                 // global_addr[31:0]
  g0[3] = (uint32_t)((ga >> 32) & 0x01ffffffu) | (2u << 30); // [120:96], type=2
  i32x8 g1;
  g1[0] = (int)(1u << 16);                              // data_size=1 (2B)
  g1[1] = (int)((tensor_d0 & 0xffffu) << 16);           // tensor_dim0[15:0]
  g1[2] = (int)((tensor_d0 >> 16) | ((tensor_d1 & 0xffffu) << 16));
  g1[3] = (int)((tensor_d1 >> 16) | (tile_d0 << 16));   // tile_dim0
  g1[4] = (int)(tile_d1 & 0xffffu);                     // tile_dim1, tile_dim2=0
  g1[5] = (int)(uint32_t)(stride0 & 0xffffffffu);       // dim0_stride[31:0]
  g1[6] = (int)(uint32_t)((stride0 >> 32) & 0xffffu);   // dim0_stride[47:32]
  g1[7] = 0;
  i32x4 z4 = {};
  i32x8 z8 = {};
  __builtin_amdgcn_tensor_load_to_lds(g0, g1, z4, z4, z8, 0);
}

// ---------------------------------------------------------------------------
// Kernel 1: projections. One wave = one 16x16 tile of one of {Q,K,V}.
// Q,K stored row-major f16 [BS][128]; V stored transposed f16 [B][128][SEQ].
// ---------------------------------------------------------------------------
__global__ __launch_bounds__(256) void
proj_f16_wmma(const float* __restrict__ x,
              const float* __restrict__ Wq,
              const float* __restrict__ Wk,
              const float* __restrict__ Wv,
              _Float16* __restrict__ qkv) {
  const int lane = threadIdx.x & 31;
  const int w    = blockIdx.x * 8 + (threadIdx.x >> 5);
  const int rowtile = w / 24;
  const int rem     = w % 24;
  const int mat     = rem >> 3;          // 0=Q 1=K 2=V
  const int coltile = rem & 7;

  const float* W = (mat == 0) ? Wq : (mat == 1 ? Wk : Wv);

  const int h  = lane >> 4;
  const int ln = lane & 15;
  const int mrow = rowtile * 16 + ln;    // flat row in [0, B*S)
  const int ncol = coltile * 16 + ln;

  v8f acc = {};
  for (int ds = 0; ds < DIN; ds += 32) {
    // A fragment (x 16x32 f16): lane holds k = 16h..16h+15
    const float* xp = x + (size_t)mrow * DIN + ds + 16 * h;
    __builtin_prefetch(xp + 32, 0, 1);
    v4f a0 = *(const v4f*)(xp);
    v4f a1 = *(const v4f*)(xp + 4);
    v4f a2 = *(const v4f*)(xp + 8);
    v4f a3 = *(const v4f*)(xp + 12);
    v16h a;
#pragma unroll
    for (int i = 0; i < 4; ++i) {
      a[i]      = (_Float16)a0[i];
      a[4 + i]  = (_Float16)a1[i];
      a[8 + i]  = (_Float16)a2[i];
      a[12 + i] = (_Float16)a3[i];
    }
    // B fragment (W 32x16): lane holds n = ncol, k = ds + 16h + i
    const float* wp = W + (size_t)(ds + 16 * h) * DOUT + ncol;
    v16h b;
#pragma unroll
    for (int i = 0; i < 16; ++i) b[i] = (_Float16)wp[(size_t)i * DOUT];

    acc = __builtin_amdgcn_wmma_f32_16x16x32_f16(false, a, false, b,
                                                 (short)0, acc, false, false);
  }
  // D layout: lane VGPR r -> element (row r + 8h, col ln)
  if (mat < 2) {
    _Float16* out = qkv + (size_t)mat * (size_t)BS * DOUT;
#pragma unroll
    for (int r = 0; r < 8; ++r)
      out[(size_t)(rowtile * 16 + r + 8 * h) * DOUT + ncol] = (_Float16)acc[r];
  } else {                                // V transposed: Vt[b][d][kv]
    _Float16* vt = qkv + (size_t)2 * BS * DOUT;
#pragma unroll
    for (int r = 0; r < 8; ++r) {
      const int row = rowtile * 16 + r + 8 * h;     // flat kv row
      const int bb  = row >> 12;                    // /SEQ
      const int rib = row & (SEQ - 1);
      vt[(size_t)bb * DOUT * SEQ + (size_t)ncol * SEQ + rib] = (_Float16)acc[r];
    }
  }
}

// ---------------------------------------------------------------------------
// Kernel 2: fused causal flash attention. One wave = 16 query rows.
// K/Vt tiles DMA'd into LDS by the TDM, double-buffered on TENSORcnt.
// ---------------------------------------------------------------------------
__global__ __launch_bounds__(128) void
flash_attn_wmma(const _Float16* __restrict__ qkv, float* __restrict__ out) {
  __shared__ alignas(128) _Float16 ldsK[4][2][32 * DOUT];   // 64 KB
  __shared__ alignas(128) _Float16 ldsV[4][2][DOUT * 32];   // 64 KB
  __shared__ alignas(16)  _Float16 pbuf[4][16 * 32];        //  4 KB

  const int lane  = threadIdx.x & 31;
  const int wv    = threadIdx.x >> 5;
  const int qtile = blockIdx.x * 4 + wv;      // 0..1023 (16 rows each)
  const int b     = qtile >> 8;               // batch
  const int qt    = qtile & 255;              // tile within batch
  const int h     = lane >> 4;
  const int ln    = lane & 15;

  const _Float16* Q   = qkv;
  const _Float16* Kb  = qkv + (size_t)BS * DOUT + (size_t)b * SEQ * DOUT;
  const _Float16* Vtb = qkv + (size_t)2 * BS * DOUT + (size_t)b * DOUT * SEQ;

  // Q as 4 A-fragments covering d = 0..127
  v16h qf[4];
  {
    const _Float16* qp = Q + (size_t)(qtile * 16 + ln) * DOUT;
#pragma unroll
    for (int f = 0; f < 4; ++f) {
      v8h r0 = *(const v8h*)(qp + f * 32 + 8 * h);
      v8h r1 = *(const v8h*)(qp + f * 32 + 16 + 8 * h);
      qf[f] = cat8(r0, r1);
    }
  }

  v8f o[8];
#pragma unroll
  for (int c = 0; c < 8; ++c) o[c] = (v8f){};
  float mrow[8], lrow[8];
#pragma unroll
  for (int r = 0; r < 8; ++r) { mrow[r] = -INFINITY; lrow[r] = 0.0f; }

  const int   qrow0 = qt * 16;
  const int   niter = ((qrow0 + 15) >> 5) + 1;      // 32-wide causal kv tiles
  const float sl    = 0.08838834764831845f * 1.4426950408889634f;

  // TDM issue for kv tile `it` into buffer `buf`
  auto issue_tiles = [&](int it, int buf) {
    const int jj = it * 32;
    // K tile: 32 rows x 128 d, row-major, contiguous rows
    tdm_load_2d(lds_off(&ldsK[wv][buf][0]), Kb + (size_t)jj * DOUT,
                DOUT, SEQ, DOUT, 32, DOUT);
    // Vt tile: 128 d-rows x 32 kv, row stride SEQ
    tdm_load_2d(lds_off(&ldsV[wv][buf][0]), Vtb + (size_t)jj,
                SEQ, DOUT, 32, DOUT, SEQ);
  };

  issue_tiles(0, 0);

  for (int it = 0; it < niter; ++it) {
    if (it + 1 < niter) {
      issue_tiles(it + 1, (it + 1) & 1);            // overlap DMA with compute
      __builtin_amdgcn_s_wait_tensorcnt(2);         // this iter's pair done
    } else {
      __builtin_amdgcn_s_wait_tensorcnt(0);
    }
    asm volatile("" ::: "memory");

    const int jj = it * 32;
    const _Float16* curK = &ldsK[wv][it & 1][0];
    const _Float16* curV = &ldsV[wv][it & 1][0];

    // S = Q @ K^T : two 16x16 score tiles
    v8f s[2];
    s[0] = (v8f){}; s[1] = (v8f){};
#pragma unroll
    for (int t = 0; t < 2; ++t) {
      const _Float16* kp = curK + (size_t)(t * 16 + ln) * DOUT;
#pragma unroll
      for (int f = 0; f < 4; ++f) {
        v8h r0 = *(const v8h*)(kp + f * 32 + 8 * h);
        v8h r1 = *(const v8h*)(kp + f * 32 + 16 + 8 * h);
        v16h kf = cat8(r0, r1);
        s[t] = __builtin_amdgcn_wmma_f32_16x16x32_f16(false, qf[f], false, kf,
                                                      (short)0, s[t], false, false);
      }
    }

    // online softmax (row r+8h lives in this lane's 16-lane half)
    float mnew[8], rsum[8];
#pragma unroll
    for (int r = 0; r < 8; ++r) {
      const int qrow = qrow0 + r + 8 * h;
      float s0 = s[0][r] * sl;
      float s1 = s[1][r] * sl;
      if (jj + ln > qrow)      s0 = -INFINITY;      // causal mask
      if (jj + 16 + ln > qrow) s1 = -INFINITY;
      float rm = fmaxf(s0, s1);
      rm = fmaxf(rm, __shfl_xor(rm, 1, 32));
      rm = fmaxf(rm, __shfl_xor(rm, 2, 32));
      rm = fmaxf(rm, __shfl_xor(rm, 4, 32));
      rm = fmaxf(rm, __shfl_xor(rm, 8, 32));
      mnew[r] = fmaxf(mrow[r], rm);
      float p0 = exp2f(s0 - mnew[r]);
      float p1 = exp2f(s1 - mnew[r]);
      s[0][r] = p0; s[1][r] = p1;
      float rs = p0 + p1;
      rs += __shfl_xor(rs, 1, 32);
      rs += __shfl_xor(rs, 2, 32);
      rs += __shfl_xor(rs, 4, 32);
      rs += __shfl_xor(rs, 8, 32);
      rsum[r] = rs;
    }
#pragma unroll
    for (int r = 0; r < 8; ++r) {
      float alpha = exp2f(mrow[r] - mnew[r]);
      mrow[r] = mnew[r];
      lrow[r] = lrow[r] * alpha + rsum[r];
#pragma unroll
      for (int c = 0; c < 8; ++c) o[c][r] *= alpha;
    }

    // transpose P from C-layout to A-layout via per-wave LDS slice
    _Float16* pb = pbuf[wv];
#pragma unroll
    for (int r = 0; r < 8; ++r) {
      pb[(r + 8 * h) * 32 + ln]      = (_Float16)s[0][r];
      pb[(r + 8 * h) * 32 + 16 + ln] = (_Float16)s[1][r];
    }
    asm volatile("s_wait_dscnt 0" ::: "memory");    // wave-local LDS RAW fence
    v8h p0 = *(const v8h*)(pb + ln * 32 + 8 * h);
    v8h p1 = *(const v8h*)(pb + ln * 32 + 16 + 8 * h);
    v16h pf = cat8(p0, p1);

    // O += P @ V : B-fragments are 8-half runs at kv 8h and 16+8h of the Vt LDS tile
#pragma unroll
    for (int c = 0; c < 8; ++c) {
      const _Float16* vp = curV + (size_t)(c * 16 + ln) * 32 + 8 * h;
      v16h vf = cat8(*(const v8h*)(vp), *(const v8h*)(vp + 16));
      o[c] = __builtin_amdgcn_wmma_f32_16x16x32_f16(false, pf, false, vf,
                                                    (short)0, o[c], false, false);
    }
  }

  // epilogue: normalize and store fp32
#pragma unroll
  for (int r = 0; r < 8; ++r) {
    float inv = 1.0f / lrow[r];
#pragma unroll
    for (int c = 0; c < 8; ++c)
      out[(size_t)(qtile * 16 + r + 8 * h) * DOUT + c * 16 + ln] = o[c][r] * inv;
  }
}

// ---------------------------------------------------------------------------
extern "C" void kernel_launch(void* const* d_in, const int* in_sizes, int n_in,
                              void* d_out, int out_size, void* d_ws, size_t ws_size,
                              hipStream_t stream) {
  (void)in_sizes; (void)n_in; (void)out_size; (void)ws_size;
  const float* x  = (const float*)d_in[0];
  const float* Wq = (const float*)d_in[1];
  const float* Wk = (const float*)d_in[2];
  const float* Wv = (const float*)d_in[3];
  float*     out  = (float*)d_out;
  _Float16*  qkv  = (_Float16*)d_ws;   // Q,K row-major + Vt transposed = 12 MB

  proj_f16_wmma<<<3072, 256, 0, stream>>>(x, Wq, Wk, Wv, qkv);
  flash_attn_wmma<<<256, 128, 0, stream>>>(qkv, out);
}
